// Net_14714557956641
// MI455X (gfx1250) — matrix operand order repaired and checked by reading, not compiled, not run on hardware
//
#include <hip/hip_runtime.h>

// ---- types for WMMA operands (gfx1250, wave32) ----
typedef _Float16 v8h  __attribute__((ext_vector_type(8)));
typedef _Float16 v16h __attribute__((ext_vector_type(16)));
typedef float    v8f  __attribute__((ext_vector_type(8)));

#define NHID   1024
#define NB     32
#define NINP   28
#define NOUT   10
#define TSTEPS 256
#define NBLK   64          // persistent grid: 64 WGs x 256 threads = 512 waves
#define KCHUNK (NHID / 4)  // K split 4 ways per output tile
#define KITER  (KCHUNK / 32)

#define BUFELEMS  (NB * NHID)       // one s or c buffer, halves
#define PINGPONG  (2 * BUFELEMS)    // distance ping->pong, halves

__device__ __constant__ const float kTwoPi    = 6.28318530717958647692f;
__device__ __constant__ const float kInvTwoPi = 0.15915494309189533577f;

// ---------------------------------------------------------------------------
// state = 0, sin(state) = 0, cos(state) = 1 (ping buffer), barrier = 0
// scbuf layout: [ s0 | c0 | s1 | c1 ], each NB*NHID halves
// ---------------------------------------------------------------------------
__global__ void k_init(float* __restrict__ state,
                       _Float16* __restrict__ scbuf,
                       unsigned* __restrict__ bar) {
    int i = blockIdx.x * blockDim.x + threadIdx.x;
    if (i < NB * NHID) {
        state[i]              = 0.0f;
        scbuf[i]              = (_Float16)0.0f;   // sin
        scbuf[i + BUFELEMS]   = (_Float16)1.0f;   // cos
    }
    if (i < 2) bar[i] = 0u;   // [0]=arrive count, [1]=generation
}

// ---------------------------------------------------------------------------
// Wh (f32 row-major [i][j]) -> f16, same layout (B-fragment rows contiguous in K)
// ---------------------------------------------------------------------------
__global__ void k_wh_to_f16(const float* __restrict__ Wh,
                            _Float16* __restrict__ Wh16) {
    int i = blockIdx.x * blockDim.x + threadIdx.x;
    if (i < NHID * NHID) Wh16[i] = (_Float16)Wh[i];
}

// ---------------------------------------------------------------------------
// Grid-wide release/acquire barrier (agent scope). All NBLK WGs co-resident.
// ---------------------------------------------------------------------------
__device__ __forceinline__ void grid_barrier(unsigned* bar, unsigned nblk) {
    __syncthreads();
    if (threadIdx.x == 0) {
        __threadfence();  // release: make this WG's stores device-visible
        unsigned g   = __hip_atomic_load(&bar[1], __ATOMIC_RELAXED,
                                         __HIP_MEMORY_SCOPE_AGENT);
        unsigned old = __hip_atomic_fetch_add(&bar[0], 1u, __ATOMIC_ACQ_REL,
                                              __HIP_MEMORY_SCOPE_AGENT);
        if (old + 1u == nblk) {
            __hip_atomic_store(&bar[0], 0u, __ATOMIC_RELAXED,
                               __HIP_MEMORY_SCOPE_AGENT);
            __hip_atomic_fetch_add(&bar[1], 1u, __ATOMIC_RELEASE,
                                   __HIP_MEMORY_SCOPE_AGENT);
        } else {
            while (__hip_atomic_load(&bar[1], __ATOMIC_ACQUIRE,
                                     __HIP_MEMORY_SCOPE_AGENT) == g) {
                __builtin_amdgcn_s_sleep(2);
            }
        }
        __threadfence();  // acquire: invalidate near caches before reading peers' data
    }
    __syncthreads();
}

// ---------------------------------------------------------------------------
// Persistent Kuramoto scan: all T=256 steps in one kernel.
//   sin(a-b) identity => per step two GEMMs [32,1024]x[1024,1024]^T via WMMA:
//     P = cos(state)*Wh^T ; Q = sin(state)*Wh^T
//     new = remainder(sin*P - cos*Q + x_t@Wi^T + Wi_b + omega + state, 2pi)
// Wave gwid in [0,512): tile = gwid>>2 (nt=tile>>1, mt=tile&1), kchunk = gwid&3.
// The wave's Wh slice (8 B-fragments) is loop-invariant but is reloaded from global
// memory every step. The sin/cos A-fragment loads are software-pipelined one
// iteration ahead so WMMAs never wait on a just-issued load. The 4 K-chunk
// waves of a tile tree-reduce partial f32 accumulators through LDS; the
// kchunk==0 wave runs the epilogue, updating state element-in-place and
// writing sin/cos f16 into the ping-pong buffer. Steps are separated by an
// atomic grid barrier.
// ---------------------------------------------------------------------------
__global__ __launch_bounds__(256) void k_scan(
    const float* __restrict__ x,          // [T, NB, NINP]
    const float* __restrict__ Wi_w,       // [NHID, NINP]
    const float* __restrict__ Wi_b,       // [NHID]
    const _Float16* __restrict__ Wh16,    // [NHID, NHID]
    const float* __restrict__ omega,      // [NHID]
    _Float16* __restrict__ scbuf,         // [ s0 | c0 | s1 | c1 ]
    float* __restrict__ state,            // [NB, NHID]
    unsigned* __restrict__ bar)
{
    __shared__ float redP[8][32][8];      // 8 KB: per-wave partial P
    __shared__ float redQ[8][32][8];      // 8 KB: per-wave partial Q

    const int lane  = threadIdx.x & 31;
    const int wslot = threadIdx.x >> 5;               // 0..7
    const int gwid  = blockIdx.x * 8 + wslot;         // 0..511
    const int tile  = gwid >> 2;                      // 0..127
    const int kch   = gwid & 3;                       // K chunk 0..3
    const int nt    = tile >> 1;                      // N tile 0..63
    const int mt    = tile & 1;                       // M tile 0..1
    const int nlo   = lane & 15;
    const int hi16  = (lane >> 4) & 1;
    const int k0    = kch * KCHUNK;

    // B-fragment (32x16 KxN): lane = column n; halves 0..7 -> K=hi16*8.., 8..15 -> +16.
    const int n = nt * 16 + nlo;
    const _Float16* whBase = Wh16 + (size_t)n * NHID + hi16 * 8 + k0;
    // A-fragment (16x32 MxK): lane = row m; halves 0..7 -> K=hi16*8.., 8..15 -> +16.
    const int    mrow  = mt * 16 + nlo;
    const size_t aIdx  = (size_t)mrow * NHID + hi16 * 8 + k0;  // pre-biased by k0
    const int    mbase = mt * 16 + hi16 * 8;          // C/D row map: VGPR r -> M

    for (int t = 0; t < TSTEPS; ++t) {
        // ping-pong via offset arithmetic on one global base (keeps loads global)
        const size_t po = (size_t)((t & 1) ? PINGPONG : 0);
        const _Float16* sPtr = scbuf + po + aIdx;
        const _Float16* cPtr = scbuf + po + BUFELEMS + aIdx;

        // prologue loads for iteration 0
        v8h clo = *(const v8h*)(cPtr);
        v8h chi = *(const v8h*)(cPtr + 16);
        v8h slo = *(const v8h*)(sPtr);
        v8h shi = *(const v8h*)(sPtr + 16);

        v8f accP = {};
        v8f accQ = {};
#pragma unroll
        for (int i = 0; i < KITER; ++i) {
            // issue next iteration's A-fragment loads before consuming current
            v8h nclo, nchi, nslo, nshi;
            if (i + 1 < KITER) {
                const int o = (i + 1) * 32;
                nclo = *(const v8h*)(cPtr + o);
                nchi = *(const v8h*)(cPtr + o + 16);
                nslo = *(const v8h*)(sPtr + o);
                nshi = *(const v8h*)(sPtr + o + 16);
            }

            v16h bfrag = __builtin_shufflevector(*(const v8h*)(whBase + i * 32), *(const v8h*)(whBase + i * 32 + 16), 0,1,2,3,4,5,6,7,8,9,10,11,12,13,14,15);  // loop-invariant across t
            v16h cfrag = __builtin_shufflevector(clo, chi, 0,1,2,3,4,5,6,7,
                                                 8,9,10,11,12,13,14,15);
            v16h sfrag = __builtin_shufflevector(slo, shi, 0,1,2,3,4,5,6,7,
                                                 8,9,10,11,12,13,14,15);
            accP = __builtin_amdgcn_wmma_f32_16x16x32_f16(
                false, cfrag, false, bfrag, (short)0, accP, false, false);
            accQ = __builtin_amdgcn_wmma_f32_16x16x32_f16(
                false, sfrag, false, bfrag, (short)0, accQ, false, false);

            if (i + 1 < KITER) { clo = nclo; chi = nchi; slo = nslo; shi = nshi; }
        }

        // ---- K-chunk reduction through LDS (4 waves -> kchunk 0 wave) ----
        if (kch != 0) {
#pragma unroll
            for (int r = 0; r < 8; ++r) {
                redP[wslot][lane][r] = accP[r];
                redQ[wslot][lane][r] = accQ[r];
            }
        }
        __syncthreads();

        if (kch == 0) {
#pragma unroll
            for (int w = 1; w < 4; ++w) {
#pragma unroll
                for (int r = 0; r < 8; ++r) {
                    accP[r] += redP[wslot + w][lane][r];
                    accQ[r] += redQ[wslot + w][lane][r];
                }
            }

            // ---- epilogue: state update for column n, rows mbase..mbase+7 ----
            _Float16* sDst = scbuf + ((t & 1) ? 0 : PINGPONG);
            _Float16* cDst = sDst + BUFELEMS;

            float wi[NINP];
#pragma unroll
            for (int i = 0; i < NINP; ++i) wi[i] = Wi_w[(size_t)n * NINP + i];
            const float addc = omega[n] + Wi_b[n];
            const float* xt  = x + (size_t)t * NB * NINP;

#pragma unroll
            for (int r = 0; r < 8; ++r) {
                const int    m   = mbase + r;
                const size_t idx = (size_t)m * NHID + n;
                const float  st  = state[idx];
                const float  sv  = __sinf(st);
                const float  cv  = __cosf(st);
                const float  coupling = sv * accP[r] - cv * accQ[r];

                float inp = 0.0f;
                const float* xr = xt + (size_t)m * NINP;
#pragma unroll
                for (int i = 0; i < NINP; ++i) inp = fmaf(xr[i], wi[i], inp);

                const float val = coupling + inp + addc + st;
                const float nw  = val - floorf(val * kInvTwoPi) * kTwoPi;
                state[idx] = nw;
                sDst[idx]  = (_Float16)__sinf(nw);
                cDst[idx]  = (_Float16)__cosf(nw);
            }
        }

        grid_barrier(bar, (unsigned)gridDim.x);
    }
}

// ---------------------------------------------------------------------------
// out[b,o] = state[b,:] . W_out[o,:] + b_out[o]
// ---------------------------------------------------------------------------
__global__ void k_readout(const float* __restrict__ state,
                          const float* __restrict__ W_out,
                          const float* __restrict__ b_out,
                          float* __restrict__ out) {
    int id = blockIdx.x * blockDim.x + threadIdx.x;
    if (id >= NB * NOUT) return;
    int b = id / NOUT, o = id % NOUT;
    float acc = b_out[o];
    const float* sr = state + (size_t)b * NHID;
    const float* wr = W_out + (size_t)o * NHID;
    for (int h = 0; h < NHID; ++h) acc = fmaf(sr[h], wr[h], acc);
    out[id] = acc;
}

// ---------------------------------------------------------------------------
extern "C" void kernel_launch(void* const* d_in, const int* in_sizes, int n_in,
                              void* d_out, int out_size, void* d_ws, size_t ws_size,
                              hipStream_t stream) {
    const float* x     = (const float*)d_in[0];   // [T, B, NINP]
    const float* Wi_w  = (const float*)d_in[1];   // [NHID, NINP]
    const float* Wi_b  = (const float*)d_in[2];   // [NHID]
    const float* Wh    = (const float*)d_in[3];   // [NHID, NHID]
    const float* omega = (const float*)d_in[4];   // [NHID]
    const float* W_out = (const float*)d_in[5];   // [NOUT, NHID]
    const float* b_out = (const float*)d_in[6];   // [NOUT]
    float* out = (float*)d_out;

    // workspace (~2.5 MB): state | Wh16 | scbuf[s0|c0|s1|c1] | barrier
    char* ws = (char*)d_ws;
    float*    state = (float*)ws;                                   // 128 KB
    _Float16* Wh16  = (_Float16*)(ws + 128 * 1024);                 // 2 MB
    _Float16* scbuf = (_Float16*)(ws + 128 * 1024 + 2 * 1024 * 1024); // 256 KB
    unsigned* bar   = (unsigned*)(scbuf + 4 * BUFELEMS);            // 2 x u32

    k_init<<<(NB * NHID + 255) / 256, 256, 0, stream>>>(state, scbuf, bar);
    k_wh_to_f16<<<(NHID * NHID + 255) / 256, 256, 0, stream>>>(Wh, Wh16);

    k_scan<<<NBLK, 256, 0, stream>>>(x, Wi_w, Wi_b, Wh16, omega,
                                     scbuf, state, bar);

    k_readout<<<(NB * NOUT + 255) / 256, 256, 0, stream>>>(state, W_out, b_out, out);
}
